// MyModel_61933428411563
// MI455X (gfx1250) — matrix operand order repaired and checked by reading, not compiled, not run on hardware
//
#include <hip/hip_runtime.h>
#include <cstdint>

// Problem constants (match reference)
#define DM     512            // d_model == INNER
#define HEADS  8
#define HD     64
#define SEQ    2048
#define BATCH  2
#define BS     (BATCH*SEQ)    // 4096 token rows
#define KT     (DM/32)        // 16 K-tiles of 32 for the projections

typedef _Float16 v16h __attribute__((ext_vector_type(16)));
typedef _Float16 v8h  __attribute__((ext_vector_type(8)));
typedef float    v8f  __attribute__((ext_vector_type(8)));

static __device__ __forceinline__ v16h cat8(v8h lo, v8h hi) {
  return __builtin_shufflevector(lo, hi, 0,1,2,3,4,5,6,7,8,9,10,11,12,13,14,15);
}

static __device__ __forceinline__ v8f wmma_f16(v16h a, v16h b, v8f c) {
  // D = A(16x32 f16) * B(32x16 f16) + C(16x16 f32)
  return __builtin_amdgcn_wmma_f32_16x16x32_f16(false, a, false, b, (short)0, c,
                                                false, false);
}

// A fragment (16x32 f16) from a row-major matrix; `base` points at [row0][k0].
// Lane (ln = lane&15, hf = lane>>4) holds K run [16*hf,16*hf+16)
// of row ln  (contiguous order, matching the B operands).
static __device__ __forceinline__ v16h load_a(const _Float16* base, int ld,
                                              int ln, int hf) {
  const _Float16* p = base + (size_t)ln * ld;
  v8h r0 = *(const v8h*)(p + hf * 16);
  v8h r1 = *(const v8h*)(p + hf * 16 + 8);
  return cat8(r0, r1);
}

// ---------------------------------------------------------------------------
// 0a. fp32 -> fp16 elementwise convert (hidden states)
// ---------------------------------------------------------------------------
__global__ void cvt_f32_f16(const float* __restrict__ src,
                            _Float16* __restrict__ dst, int n) {
  int i = blockIdx.x * blockDim.x + threadIdx.x;
  if (i < n) dst[i] = (_Float16)src[i];
}

// ---------------------------------------------------------------------------
// 0b. Pack a [512 x N] f32 weight (row-major, rows = K) into the per-lane
// B-fragment layout: out[((nt*KT + kt)*32 + lane)*16 + j] = W[k][n]
//   n  = nt*16 + (lane&15)
//   k  = kt*32 + (lane>>4)*16 + j          (contiguous k order, matching load_a)
// So in the GEMM each lane loads its 16 halves with one contiguous 32B load.
// ---------------------------------------------------------------------------
__global__ void pack_w(const float* __restrict__ W, _Float16* __restrict__ out,
                       int N) {
  int e = blockIdx.x * blockDim.x + threadIdx.x;
  if (e >= DM * N) return;
  int j    = e & 15;
  int lane = (e >> 4) & 31;
  int kt   = (e >> 9) & (KT - 1);
  int nt   = e >> 13;
  int k = kt * 32 + (lane >> 4) * 16 + j;
  int n = nt * 16 + (lane & 15);
  out[e] = (_Float16)W[(size_t)k * N + n];
}

// ---------------------------------------------------------------------------
// 1. Fused Q/K/V projection. One wave computes a 16-token x 64-col (one head)
// tile for all three matrices: A fragment reused across 12 WMMAs per K-step.
// q,k stored [b,h,s,64] row-major; v stored transposed [b,h,64,s].
// ---------------------------------------------------------------------------
__global__ __launch_bounds__(256)
void qkv_proj(const _Float16* __restrict__ h16,
              const _Float16* __restrict__ wqp,
              const _Float16* __restrict__ wkp,
              const _Float16* __restrict__ wvp,
              _Float16* __restrict__ q16,
              _Float16* __restrict__ k16,
              _Float16* __restrict__ vT16) {
  const int lane = threadIdx.x & 31;
  const int wv   = threadIdx.x >> 5;
  const int ln   = lane & 15, hf = lane >> 4;
  const int row0 = blockIdx.x * 128 + wv * 16;   // token row tile
  const int h    = blockIdx.y;                   // head

  const v8f zero = {0.f,0.f,0.f,0.f,0.f,0.f,0.f,0.f};
  v8f accQ[4], accK[4], accV[4];
#pragma unroll
  for (int t = 0; t < 4; ++t) { accQ[t] = zero; accK[t] = zero; accV[t] = zero; }

  for (int kt = 0; kt < KT; ++kt) {
    v16h a = load_a(h16 + (size_t)row0 * DM + kt * 32, DM, ln, hf);
#pragma unroll
    for (int t = 0; t < 4; ++t) {
      const int nt = h * 4 + t;
      const size_t boff = ((size_t)(nt * KT + kt) * 32 + lane) * 16;
      v16h bq = *(const v16h*)(wqp + boff);
      accQ[t] = wmma_f16(a, bq, accQ[t]);
      v16h bk = *(const v16h*)(wkp + boff);
      accK[t] = wmma_f16(a, bk, accK[t]);
      v16h bv = *(const v16h*)(wvp + boff);
      accV[t] = wmma_f16(a, bv, accV[t]);
    }
  }

  // C layout: VGPR r -> row (r + 8*hf), col = ln within the 16-wide tile.
#pragma unroll
  for (int t = 0; t < 4; ++t) {
#pragma unroll
    for (int r = 0; r < 8; ++r) {
      const int tok = row0 + r + 8 * hf;
      const int b = tok >> 11, s = tok & (SEQ - 1);
      const int d = t * 16 + ln;
      const size_t qk = (((size_t)(b * HEADS + h)) * SEQ + s) * HD + d;
      q16[qk] = (_Float16)accQ[t][r];
      k16[qk] = (_Float16)accK[t][r];
      vT16[(((size_t)(b * HEADS + h)) * HD + d) * SEQ + s] = (_Float16)accV[t][r];
    }
  }
}

// ---------------------------------------------------------------------------
// 2. Flash attention: one wave owns (b,h, 16 queries); streams 32-key chunks.
// scores = Q*0.125 @ K^T (2 N-tiles x 2 K-steps = 4 WMMA), online softmax,
// P routed through per-wave LDS tile (C-layout -> A-layout), PV = 4 WMMA.
// Output written f16 to attn16 [b*S+s][512].
// ---------------------------------------------------------------------------
__global__ __launch_bounds__(256)
void attn_fwd(const _Float16* __restrict__ q16,
              const _Float16* __restrict__ k16,
              const _Float16* __restrict__ vT16,
              _Float16* __restrict__ attn16) {
  __shared__ _Float16 plds[8][16 * 32];   // 1KB P tile per wave

  const int lane = threadIdx.x & 31;
  const int wv   = threadIdx.x >> 5;
  const int ln   = lane & 15, hf = lane >> 4;
  const int bh = blockIdx.x;
  const int b  = bh >> 3, h = bh & 7;
  const int q0 = (blockIdx.y * 8 + wv) * 16;

  const _Float16* Qh = q16  + (size_t)bh * SEQ * HD;
  const _Float16* Kh = k16  + (size_t)bh * SEQ * HD;
  const _Float16* Vh = vT16 + (size_t)bh * HD * SEQ;

  // Q fragments (head_dim split 0..31 / 32..63), pre-scaled by 1/sqrt(64)
  v16h qa[2];
#pragma unroll
  for (int dc = 0; dc < 2; ++dc) {
    qa[dc] = load_a(Qh + (size_t)q0 * HD + dc * 32, HD, ln, hf);
#pragma unroll
    for (int i = 0; i < 16; ++i) qa[dc][i] = qa[dc][i] * (_Float16)0.125f;
  }

  const v8f zero = {0.f,0.f,0.f,0.f,0.f,0.f,0.f,0.f};
  float mrow[8], lrow[8];
  v8f o[4];
#pragma unroll
  for (int r = 0; r < 8; ++r) { mrow[r] = -3.0e38f; lrow[r] = 0.f; }
#pragma unroll
  for (int t = 0; t < 4; ++t) o[t] = zero;

  for (int kb = 0; kb < SEQ / 32; ++kb) {
    const int keyB = kb * 32;
    if (kb + 1 < SEQ / 32) {   // global_prefetch_b8 next K rows / V columns
      __builtin_prefetch(Kh + (size_t)(keyB + 32 + lane) * HD, 0, 1);
      __builtin_prefetch(Vh + (size_t)lane * SEQ + keyB + 32, 0, 1);
      __builtin_prefetch(Vh + (size_t)(lane + 32) * SEQ + keyB + 32, 0, 1);
    }

    // --- scores: two 16-key column tiles, K-dim = head dim (contiguous) ---
    v8f c0 = zero, c1 = zero;
#pragma unroll
    for (int dc = 0; dc < 2; ++dc) {
      v16h b0 = *(const v16h*)(Kh + (size_t)(keyB + ln) * HD + dc * 32 + hf * 16);
      c0 = wmma_f16(qa[dc], b0, c0);
      v16h b1 = *(const v16h*)(Kh + (size_t)(keyB + 16 + ln) * HD + dc * 32 + hf * 16);
      c1 = wmma_f16(qa[dc], b1, c1);
    }

    // --- online softmax (rows r+8*hf live across 16-lane halves) ---
    float rmax[8], rsum[8], alpha[8];
#pragma unroll
    for (int r = 0; r < 8; ++r) rmax[r] = fmaxf(c0[r], c1[r]);
#pragma unroll
    for (int m = 1; m <= 8; m <<= 1)
#pragma unroll
      for (int r = 0; r < 8; ++r)
        rmax[r] = fmaxf(rmax[r], __shfl_xor(rmax[r], m, 32));
#pragma unroll
    for (int r = 0; r < 8; ++r) {
      const float mn = fmaxf(mrow[r], rmax[r]);
      alpha[r] = __expf(mrow[r] - mn);
      mrow[r] = mn;
      const float p0 = __expf(c0[r] - mn);
      const float p1 = __expf(c1[r] - mn);
      c0[r] = p0; c1[r] = p1;
      rsum[r] = p0 + p1;
    }
#pragma unroll
    for (int m = 1; m <= 8; m <<= 1)
#pragma unroll
      for (int r = 0; r < 8; ++r) rsum[r] += __shfl_xor(rsum[r], m, 32);
#pragma unroll
    for (int r = 0; r < 8; ++r) lrow[r] = lrow[r] * alpha[r] + rsum[r];
#pragma unroll
    for (int t = 0; t < 4; ++t)
#pragma unroll
      for (int r = 0; r < 8; ++r) o[t][r] = o[t][r] * alpha[r];

    // --- P: C-layout -> LDS -> A-fragment layout ---
    _Float16* pw = &plds[wv][0];
#pragma unroll
    for (int r = 0; r < 8; ++r) {
      const int row = r + 8 * hf;
      pw[row * 32 + ln]      = (_Float16)c0[r];
      pw[row * 32 + 16 + ln] = (_Float16)c1[r];
    }
    asm volatile("s_wait_dscnt 0" ::: "memory");
    const _Float16* pr = pw + ln * 32;
    v8h p0 = *(const v8h*)(pr + hf * 16);
    v8h p1 = *(const v8h*)(pr + hf * 16 + 8);
    v16h pa = cat8(p0, p1);

    // --- O += P @ V  (V transposed: WMMA-K = key index, contiguous) ---
#pragma unroll
    for (int t = 0; t < 4; ++t) {
      v16h bv = *(const v16h*)(Vh + (size_t)(t * 16 + ln) * SEQ + keyB + hf * 16);
      o[t] = wmma_f16(pa, bv, o[t]);
    }
  }

  float inv[8];
#pragma unroll
  for (int r = 0; r < 8; ++r) inv[r] = 1.0f / lrow[r];
#pragma unroll
  for (int t = 0; t < 4; ++t)
#pragma unroll
    for (int r = 0; r < 8; ++r) {
      const int srow = q0 + r + 8 * hf;
      attn16[((size_t)(b * SEQ + srow)) * DM + h * HD + t * 16 + ln] =
          (_Float16)(o[t][r] * inv[r]);
    }
}

// ---------------------------------------------------------------------------
// 3. Output projection: out(f32) = attn16 @ Wo  (wave = 16 rows x 64 cols)
// ---------------------------------------------------------------------------
__global__ __launch_bounds__(256)
void out_proj(const _Float16* __restrict__ a16,
              const _Float16* __restrict__ wop,
              float* __restrict__ out) {
  const int lane = threadIdx.x & 31;
  const int wv   = threadIdx.x >> 5;
  const int ln   = lane & 15, hf = lane >> 4;
  const int row0 = blockIdx.x * 128 + wv * 16;
  const int nt0  = blockIdx.y * 4;

  const v8f zero = {0.f,0.f,0.f,0.f,0.f,0.f,0.f,0.f};
  v8f acc[4];
#pragma unroll
  for (int t = 0; t < 4; ++t) acc[t] = zero;

  for (int kt = 0; kt < KT; ++kt) {
    v16h a = load_a(a16 + (size_t)row0 * DM + kt * 32, DM, ln, hf);
#pragma unroll
    for (int t = 0; t < 4; ++t) {
      const size_t boff = ((size_t)((nt0 + t) * KT + kt) * 32 + lane) * 16;
      v16h bw = *(const v16h*)(wop + boff);
      acc[t] = wmma_f16(a, bw, acc[t]);
    }
  }
#pragma unroll
  for (int t = 0; t < 4; ++t)
#pragma unroll
    for (int r = 0; r < 8; ++r)
      out[(size_t)(row0 + r + 8 * hf) * DM + (nt0 + t) * 16 + ln] = acc[t][r];
}

// ---------------------------------------------------------------------------
// Launch.  Workspace layout (f16, ~22 MiB total):
//   [0,4M)   h16          [4M,4.5M) Wq packed   [4.5M,5M) Wk packed
//   [5M,5.5M) Wv packed   [5.5M,6M) Wo packed
//   [6M,10M) q16 [b,h,s,64]   [10M,14M) k16 [b,h,s,64]
//   [14M,18M) vT16 [b,h,64,s] [18M,22M) attn16 [b*s,512]
// ---------------------------------------------------------------------------
extern "C" void kernel_launch(void* const* d_in, const int* in_sizes, int n_in,
                              void* d_out, int out_size, void* d_ws, size_t ws_size,
                              hipStream_t stream) {
  (void)in_sizes; (void)n_in; (void)out_size; (void)ws_size;
  const float* hs = (const float*)d_in[0];
  const float* Wq = (const float*)d_in[1];
  const float* Wk = (const float*)d_in[2];
  const float* Wv = (const float*)d_in[3];
  const float* Wo = (const float*)d_in[4];

  char* ws = (char*)d_ws;
  const size_t MB = (size_t)1 << 20;
  _Float16* h16  = (_Float16*)(ws + 0);
  _Float16* wqp  = (_Float16*)(ws + 4 * MB);
  _Float16* wkp  = (_Float16*)(ws + 4 * MB + 512 * 1024);
  _Float16* wvp  = (_Float16*)(ws + 5 * MB);
  _Float16* wop  = (_Float16*)(ws + 5 * MB + 512 * 1024);
  _Float16* q16  = (_Float16*)(ws + 6 * MB);
  _Float16* k16  = (_Float16*)(ws + 10 * MB);
  _Float16* vT16 = (_Float16*)(ws + 14 * MB);
  _Float16* a16  = (_Float16*)(ws + 18 * MB);

  cvt_f32_f16<<<(BS * DM + 255) / 256, 256, 0, stream>>>(hs, h16, BS * DM);
  pack_w<<<(DM * DM + 255) / 256, 256, 0, stream>>>(Wq, wqp, DM);
  pack_w<<<(DM * DM + 255) / 256, 256, 0, stream>>>(Wk, wkp, DM);
  pack_w<<<(DM * DM + 255) / 256, 256, 0, stream>>>(Wv, wvp, DM);
  pack_w<<<(DM * DM + 255) / 256, 256, 0, stream>>>(Wo, wop, DM);

  qkv_proj<<<dim3(BS / 128, HEADS), 256, 0, stream>>>(h16, wqp, wkp, wvp,
                                                      q16, k16, vT16);
  attn_fwd<<<dim3(BATCH * HEADS, SEQ / 16 / 8), 256, 0, stream>>>(q16, k16,
                                                                  vT16, a16);
  out_proj<<<dim3(BS / 128, DM / 64), 256, 0, stream>>>(a16, wop,
                                                        (float*)d_out);
}
